// Pyramid_49701361550134
// MI455X (gfx1250) — matrix operand order repaired and checked by reading, not compiled, not run on hardware
//
#include <hip/hip_runtime.h>
#include <hip/hip_bf16.h>

// ---------------------------------------------------------------------------
// Types / WMMA helpers (CDNA5 gfx1250, wave32)
// ---------------------------------------------------------------------------
typedef __bf16 bf16_t;
typedef __attribute__((ext_vector_type(16))) __bf16 v16bf;
typedef __attribute__((ext_vector_type(8)))  __bf16 v8bf;
typedef __attribute__((ext_vector_type(8)))  float   v8f;

#define WMMA_BF16(A_, B_, C_) \
  __builtin_amdgcn_wmma_f32_16x16x32_bf16(false, (A_), false, (B_), (short)0, (C_), false, false)

__device__ __forceinline__ float sigf(float x) { return 1.0f / (1.0f + __expf(-x)); }
__device__ __forceinline__ float tanh_fast(float x) { return 2.0f * sigf(2.0f * x) - 1.0f; }

// A fragment: 16x32 bf16.  lane&15 = row M. Per ISA layout:
//   lanes 0-15 : e=0..7 -> K=k0+e,      e=8..15 -> K=k0+16+(e-8)
//   lanes 16-31: e=0..7 -> K=k0+8+e,    e=8..15 -> K=k0+24+(e-8)
__device__ __forceinline__ v16bf load_a_frag(const bf16_t* row, int k0, int hi) {
  v8bf g1 = *(const v8bf*)(row + k0 + 8 * hi);
  v8bf g2 = *(const v8bf*)(row + k0 + 16 + 8 * hi);
  v16bf a;
#pragma unroll
  for (int e = 0; e < 8; ++e) { a[e] = g1[e]; a[e + 8] = g2[e]; }
  return a;
}

// B fragment: 32x16 bf16.  lane&15 = column N; e=0..7 -> K=k0+8*hi+e, e=8..15 -> K=k0+16+8*hi+(e-8).
__device__ __forceinline__ v16bf load_b_frag(const bf16_t* B, int ldb, int k0, int n0, int lane) {
  const bf16_t* col = B + n0 + (lane & 15);
  const int hb = lane >> 4;
  v16bf b;
#pragma unroll
  for (int e = 0; e < 8; ++e) {
    b[e] = col[(size_t)(k0 + 8 * hb + e) * ldb];
    b[e + 8] = col[(size_t)(k0 + 16 + 8 * hb + e) * ldb];
  }
  return b;
}

// ---------------------------------------------------------------------------
// Problem constants
// ---------------------------------------------------------------------------
#define PB_BATCH 128
#define PB_SEQ   64
#define PB_WDIM  300
#define PB_WDIMP 320     // padded K for projection
#define PB_MDIM  512
#define PB_HID   256
#define PB_GATES 1280    // 5*HID
#define PB_MLP   1024
#define PB_NCLS  3

// ---------------------------------------------------------------------------
// Prep kernels (f32 -> bf16 conversions, gathers, packing)
// ---------------------------------------------------------------------------
__global__ void gather_embed_kernel(bf16_t* __restrict__ dst,
                                    const float* __restrict__ table,
                                    const int* __restrict__ sents) {
  int idx = blockIdx.x * blockDim.x + threadIdx.x;
  const int total = PB_BATCH * PB_SEQ * PB_WDIMP;
  if (idx >= total) return;
  int r = idx / PB_WDIMP;
  int k = idx - r * PB_WDIMP;
  float v = (k < PB_WDIM) ? table[(size_t)sents[r] * PB_WDIM + k] : 0.0f;
  dst[idx] = (bf16_t)v;
}

__global__ void pad_wproj_kernel(bf16_t* __restrict__ dst, const float* __restrict__ src) {
  int idx = blockIdx.x * blockDim.x + threadIdx.x;
  if (idx >= PB_WDIMP * PB_MDIM) return;
  int k = idx / PB_MDIM;
  int n = idx - k * PB_MDIM;
  dst[idx] = (bf16_t)((k < PB_WDIM) ? src[k * PB_MDIM + n] : 0.0f);
}

__global__ void concat_wlr_kernel(bf16_t* __restrict__ dst,
                                  const float* __restrict__ Wl,
                                  const float* __restrict__ Wr) {
  int idx = blockIdx.x * blockDim.x + threadIdx.x;
  if (idx >= 2 * PB_HID * PB_GATES) return;
  int k = idx / PB_GATES;
  int n = idx - k * PB_GATES;
  float v = (k < PB_HID) ? Wl[k * PB_GATES + n] : Wr[(k - PB_HID) * PB_GATES + n];
  dst[idx] = (bf16_t)v;
}

__global__ void cvt_f32_bf16_kernel(bf16_t* __restrict__ dst, const float* __restrict__ src, int n) {
  int idx = blockIdx.x * blockDim.x + threadIdx.x;
  if (idx < n) dst[idx] = (bf16_t)src[idx];
}

// Pack final [h | c] (position 0) into dense 128x512 bf16 matrix for the MLP.
__global__ void pack_final_kernel(bf16_t* __restrict__ X0,
                                  const bf16_t* __restrict__ Hf,
                                  const float* __restrict__ Cf) {
  int idx = blockIdx.x * blockDim.x + threadIdx.x;
  if (idx >= PB_BATCH * PB_MDIM) return;
  int b = idx >> 9;          // /512
  int k = idx & 511;
  size_t row = (size_t)b * PB_SEQ * PB_HID;   // position 0
  float v = (k < PB_HID) ? (float)Hf[row + k] : Cf[row + (k - PB_HID)];
  X0[idx] = (bf16_t)v;
}

// Tiny output layer: (128x1024 f32) @ (1024x3) + bias
__global__ void final_out_kernel(float* __restrict__ out,
                                 const float* __restrict__ X,
                                 const float* __restrict__ Wout,
                                 const float* __restrict__ bout) {
  int idx = blockIdx.x * blockDim.x + threadIdx.x;
  if (idx >= PB_BATCH * PB_NCLS) return;
  int m = idx / PB_NCLS;
  int c = idx - m * PB_NCLS;
  float s = bout[c];
  const float* xr = X + (size_t)m * PB_MLP;
  for (int k = 0; k < PB_MLP; ++k) s += xr[k] * Wout[k * PB_NCLS + c];
  out[idx] = s;
}

// ---------------------------------------------------------------------------
// Generic bf16 WMMA GEMM: each wave computes a 16(M) x 64(N) tile.
// MODE 1: bias + relu, write bf16 out + f32 out (ldo)
// MODE 2: projection epilogue: bias, n<256 -> Hbuf (bf16), n>=256 -> Cbuf (f32)
// M % 16 == 0, N % 64 == 0, K % 32 == 0 required.
// ---------------------------------------------------------------------------
template <int MODE>
__global__ __launch_bounds__(256, 1) void gemm_bf16_kernel(
                                 const bf16_t* __restrict__ A, int lda,
                                 const bf16_t* __restrict__ B, int ldb,
                                 const float* __restrict__ bias,
                                 int M, int K, int N,
                                 bf16_t* __restrict__ outBf,
                                 float* __restrict__ outF, int ldo) {
  const int lane = threadIdx.x & 31;
  const int wid = (blockIdx.x * blockDim.x + threadIdx.x) >> 5;
  const int tilesM = M >> 4;
  const int tilesN = N >> 6;
  if (wid >= tilesM * tilesN) return;
  const int mt = wid / tilesN;
  const int nt = wid - mt * tilesN;
  const int n0 = nt * 64;
  const int laneLo = lane & 15;
  const int hi = lane >> 4;

  const bf16_t* arow = A + (size_t)(mt * 16 + laneLo) * lda;
  v8f acc[4] = {v8f{}, v8f{}, v8f{}, v8f{}};

  for (int k0 = 0; k0 < K; k0 += 32) {
    v16bf af = load_a_frag(arow, k0, hi);
#pragma unroll
    for (int t = 0; t < 4; ++t) {
      v16bf bf = load_b_frag(B, ldb, k0, n0 + t * 16, lane);
      acc[t] = WMMA_BF16(af, bf, acc[t]);
    }
  }

#pragma unroll
  for (int r = 0; r < 8; ++r) {
    int m = mt * 16 + r + 8 * hi;
#pragma unroll
    for (int t = 0; t < 4; ++t) {
      int n = n0 + t * 16 + laneLo;
      float v = acc[t][r] + bias[n];
      if (MODE == 1) {
        v = fmaxf(v, 0.0f);
        outBf[(size_t)m * ldo + n] = (bf16_t)v;
        outF[(size_t)m * ldo + n] = v;
      } else {  // MODE == 2 : split projection output into H (bf16) / C (f32)
        if (n < PB_HID)
          outBf[(size_t)m * PB_HID + n] = (bf16_t)v;
        else
          outF[(size_t)m * PB_HID + (n - PB_HID)] = v;
      }
    }
  }
}

// ---------------------------------------------------------------------------
// Fused tree-LSTM level: one wave = 32 rows x 16 cols of h/c output,
// all 5 gate blocks (i, fl, fr, o, g): 10 WMMA accumulators, every B
// fragment reused for two A tiles (1.4 loads per WMMA instead of 2.4).
// __launch_bounds__(256,1) raises the VGPR budget so the 80 accumulator
// VGPRs + pipelined fragments stay in registers (no scratch spills);
// K-loop unroll capped at 2 to bound in-flight fragment count.
// A = [h_l | h_r] is 512 contiguous bf16 at H[(b*64+j)*256] (h-only storage!).
// M = 128*Lwm1 is always a multiple of 32.
// ---------------------------------------------------------------------------
__global__ __launch_bounds__(256, 1) void tree_level_kernel(
                                  const bf16_t* __restrict__ Hin,
                                  const float* __restrict__ Cin,
                                  bf16_t* __restrict__ Hout,
                                  float* __restrict__ Cout,
                                  const bf16_t* __restrict__ W,     // 512 x 1280 bf16
                                  const float* __restrict__ bcomp,  // 1280
                                  int Lwm1) {
  const int lane = threadIdx.x & 31;
  const int wid = (blockIdx.x * blockDim.x + threadIdx.x) >> 5;
  const int tilesM2 = 4 * Lwm1;  // M/32 with M = 128*Lwm1
  const int mt = wid >> 4;       // 16 N-strips over HID=256
  const int nt = wid & 15;
  if (mt >= tilesM2) return;
  const int laneLo = lane & 15;
  const int hi = lane >> 4;
  const int n0 = nt * 16;
  const int mbase = mt * 32;

  // Per-lane A row pointers for the two 16-row sub-tiles.
  int mA0 = mbase + laneLo;
  int b0 = mA0 / Lwm1;
  int j0 = mA0 - b0 * Lwm1;
  const bf16_t* arow0 = Hin + (size_t)(b0 * PB_SEQ + j0) * PB_HID;
  int mA1 = mbase + 16 + laneLo;
  int b1 = mA1 / Lwm1;
  int j1 = mA1 - b1 * Lwm1;
  const bf16_t* arow1 = Hin + (size_t)(b1 * PB_SEQ + j1) * PB_HID;

  v8f acc0[5] = {v8f{}, v8f{}, v8f{}, v8f{}, v8f{}};
  v8f acc1[5] = {v8f{}, v8f{}, v8f{}, v8f{}, v8f{}};

#pragma unroll 2
  for (int k0 = 0; k0 < 2 * PB_HID; k0 += 32) {
    v16bf afA = load_a_frag(arow0, k0, hi);
    v16bf afB = load_a_frag(arow1, k0, hi);
#pragma unroll
    for (int g = 0; g < 5; ++g) {
      v16bf bf = load_b_frag(W, PB_GATES, k0, n0 + g * PB_HID, lane);
      acc0[g] = WMMA_BF16(afA, bf, acc0[g]);
      acc1[g] = WMMA_BF16(afB, bf, acc1[g]);
    }
  }

  const int n = n0 + laneLo;
  const float bi = bcomp[n];
  const float bfl = bcomp[PB_HID + n];
  const float bfr = bcomp[2 * PB_HID + n];
  const float bo = bcomp[3 * PB_HID + n];
  const float bg = bcomp[4 * PB_HID + n];

#pragma unroll
  for (int sub = 0; sub < 2; ++sub) {
#pragma unroll
    for (int r = 0; r < 8; ++r) {
      int m = mbase + sub * 16 + r + 8 * hi;
      int b = m / Lwm1;
      int j = m - b * Lwm1;
      size_t rowL = (size_t)(b * PB_SEQ + j) * PB_HID;
      float cl = Cin[rowL + n];
      float cr = Cin[rowL + PB_HID + n];  // position j+1 is contiguous
      float gi = (sub ? acc1[0][r] : acc0[0][r]) + bi;
      float gl = (sub ? acc1[1][r] : acc0[1][r]) + bfl;
      float gr = (sub ? acc1[2][r] : acc0[2][r]) + bfr;
      float go = (sub ? acc1[3][r] : acc0[3][r]) + bo;
      float gg = (sub ? acc1[4][r] : acc0[4][r]) + bg;
      float c = cl * sigf(gl) + cr * sigf(gr) + sigf(gi) * tanh_fast(gg);
      float h = sigf(go) * tanh_fast(c);
      Hout[rowL + n] = (bf16_t)h;
      Cout[rowL + n] = c;
    }
  }
}

// ---------------------------------------------------------------------------
// Host-side launch
// ---------------------------------------------------------------------------
static inline size_t alignUp256(size_t x) { return (x + 255) & ~(size_t)255; }

extern "C" void kernel_launch(void* const* d_in, const int* in_sizes, int n_in,
                              void* d_out, int out_size, void* d_ws, size_t ws_size,
                              hipStream_t stream) {
  (void)in_sizes; (void)n_in; (void)out_size; (void)ws_size;

  const int*   sents   = (const int*)d_in[0];
  // d_in[1] = transitions (unused)
  const float* table   = (const float*)d_in[2];
  const float* W_proj  = (const float*)d_in[3];
  const float* b_proj  = (const float*)d_in[4];
  const float* W_l     = (const float*)d_in[5];
  const float* W_r     = (const float*)d_in[6];
  const float* b_comp  = (const float*)d_in[7];
  const float* mlp_W0  = (const float*)d_in[8];
  const float* mlp_b0  = (const float*)d_in[9];
  const float* mlp_W1  = (const float*)d_in[10];
  const float* mlp_b1  = (const float*)d_in[11];
  const float* mlp_Wout = (const float*)d_in[12];
  const float* mlp_bout = (const float*)d_in[13];
  float* out = (float*)d_out;

  // Workspace carve-out (256B aligned slices)
  char* p = (char*)d_ws;
  auto carve = [&](size_t bytes) { char* q = p; p += alignUp256(bytes); return (void*)q; };

  const size_t stateElems = (size_t)PB_BATCH * PB_SEQ * PB_HID;  // 2M
  bf16_t* H0   = (bf16_t*)carve(stateElems * sizeof(bf16_t));
  bf16_t* H1   = (bf16_t*)carve(stateElems * sizeof(bf16_t));
  float*  C0   = (float*)carve(stateElems * sizeof(float));
  float*  C1   = (float*)carve(stateElems * sizeof(float));
  bf16_t* Aemb = (bf16_t*)carve((size_t)PB_BATCH * PB_SEQ * PB_WDIMP * sizeof(bf16_t));
  bf16_t* Wp   = (bf16_t*)carve((size_t)PB_WDIMP * PB_MDIM * sizeof(bf16_t));
  bf16_t* Wlr  = (bf16_t*)carve((size_t)2 * PB_HID * PB_GATES * sizeof(bf16_t));
  bf16_t* W0b  = (bf16_t*)carve((size_t)PB_MDIM * PB_MLP * sizeof(bf16_t));
  bf16_t* W1b  = (bf16_t*)carve((size_t)PB_MLP * PB_MLP * sizeof(bf16_t));
  bf16_t* X0   = (bf16_t*)carve((size_t)PB_BATCH * PB_MDIM * sizeof(bf16_t));
  bf16_t* X1b  = (bf16_t*)carve((size_t)PB_BATCH * PB_MLP * sizeof(bf16_t));
  float*  X1f  = (float*)carve((size_t)PB_BATCH * PB_MLP * sizeof(float));
  bf16_t* X2b  = (bf16_t*)carve((size_t)PB_BATCH * PB_MLP * sizeof(bf16_t));
  float*  X2f  = (float*)carve((size_t)PB_BATCH * PB_MLP * sizeof(float));

  const int TPB = 256;  // 8 waves / block

  // --- weight / input conversions -----------------------------------------
  {
    int n = PB_BATCH * PB_SEQ * PB_WDIMP;
    gather_embed_kernel<<<(n + TPB - 1) / TPB, TPB, 0, stream>>>(Aemb, table, sents);
  }
  {
    int n = PB_WDIMP * PB_MDIM;
    pad_wproj_kernel<<<(n + TPB - 1) / TPB, TPB, 0, stream>>>(Wp, W_proj);
  }
  {
    int n = 2 * PB_HID * PB_GATES;
    concat_wlr_kernel<<<(n + TPB - 1) / TPB, TPB, 0, stream>>>(Wlr, W_l, W_r);
  }
  {
    int n = PB_MDIM * PB_MLP;
    cvt_f32_bf16_kernel<<<(n + TPB - 1) / TPB, TPB, 0, stream>>>(W0b, mlp_W0, n);
  }
  {
    int n = PB_MLP * PB_MLP;
    cvt_f32_bf16_kernel<<<(n + TPB - 1) / TPB, TPB, 0, stream>>>(W1b, mlp_W1, n);
  }

  // --- projection: (8192 x 320) @ (320 x 512) -> split H (bf16) / C (f32) --
  {
    int M = PB_BATCH * PB_SEQ;              // 8192
    int waves = (M >> 4) * (PB_MDIM >> 6);  // 512 * 8 = 4096
    int blocks = (waves + 7) / 8;
    gemm_bf16_kernel<2><<<blocks, TPB, 0, stream>>>(Aemb, PB_WDIMP, Wp, PB_MDIM,
                                                    b_proj, M, PB_WDIMP, PB_MDIM,
                                                    H0, C0, PB_HID);
  }

  // --- 63 tree-LSTM levels (ping-pong) -------------------------------------
  bf16_t* Hcur = H0; float* Ccur = C0;
  bf16_t* Hnxt = H1; float* Cnxt = C1;
  for (int l = 0; l < PB_SEQ - 1; ++l) {
    int Lwm1 = (PB_SEQ - l) - 1;            // output width
    int tilesM2 = 4 * Lwm1;                 // M/32 with M = 128*Lwm1
    int waves = tilesM2 * 16;               // 16 N-strips
    int blocks = (waves + 7) / 8;
    tree_level_kernel<<<blocks, TPB, 0, stream>>>(Hcur, Ccur, Hnxt, Cnxt, Wlr, b_comp, Lwm1);
    bf16_t* th = Hcur; Hcur = Hnxt; Hnxt = th;
    float*  tc = Ccur; Ccur = Cnxt; Cnxt = tc;
  }

  // --- MLP head -------------------------------------------------------------
  {
    int n = PB_BATCH * PB_MDIM;
    pack_final_kernel<<<(n + TPB - 1) / TPB, TPB, 0, stream>>>(X0, Hcur, Ccur);
  }
  {
    int waves = (PB_BATCH >> 4) * (PB_MLP >> 6);  // 8 * 16 = 128
    int blocks = (waves + 7) / 8;
    gemm_bf16_kernel<1><<<blocks, TPB, 0, stream>>>(X0, PB_MDIM, W0b, PB_MLP,
                                                    mlp_b0, PB_BATCH, PB_MDIM, PB_MLP,
                                                    X1b, X1f, PB_MLP);
    gemm_bf16_kernel<1><<<blocks, TPB, 0, stream>>>(X1b, PB_MLP, W1b, PB_MLP,
                                                    mlp_b1, PB_BATCH, PB_MLP, PB_MLP,
                                                    X2b, X2f, PB_MLP);
  }
  {
    int n = PB_BATCH * PB_NCLS;  // 384
    final_out_kernel<<<(n + TPB - 1) / TPB, TPB, 0, stream>>>(out, X2f, mlp_Wout, mlp_bout);
  }
}
